// PatchMatcher_58909771432259
// MI455X (gfx1250) — matrix operand order repaired and checked by reading, not compiled, not run on hardware
//
#include <hip/hip_runtime.h>
#include <hip/hip_bf16.h>
#include <math.h>

// Problem constants (from reference setup_inputs)
#define BATCH 2
#define CCH   256      // feature channels
#define MCH   4        // mask channels
#define HH    64
#define WW    64
#define NPIX  4096     // HH*WW
#define KDIM  256      // GEMM K == CCH
#define EPSV  1e-12f

typedef __attribute__((ext_vector_type(16))) _Float16 v16h;
typedef __attribute__((ext_vector_type(8)))  float    v8f;

union FragU { v16h v; uint4 u[2]; };

// ---------------------------------------------------------------------------
// Kernel 1: transpose features to [b*N + n][c] fp16 and per-pixel sq-norms
// ---------------------------------------------------------------------------
__global__ __launch_bounds__(256)
void prep_feat_kernel(const float* __restrict__ feat, _Float16* __restrict__ Xt,
                      float* __restrict__ sq) {
    int idx = blockIdx.x * blockDim.x + threadIdx.x;   // over BATCH*NPIX
    if (idx >= BATCH * NPIX) return;
    int b = idx >> 12;
    int n = idx & (NPIX - 1);
    const float* src = feat + (size_t)b * CCH * NPIX + n;
    _Float16* dst = Xt + (size_t)idx * KDIM;
    float s = 0.f;
#pragma unroll 4
    for (int c = 0; c < CCH; ++c) {
        float v = src[(size_t)c * NPIX];   // coalesced across threads (adjacent n)
        dst[c] = (_Float16)v;
        s += v * v;
    }
    sq[idx] = s;
}

// ---------------------------------------------------------------------------
// Kernel 2: pack 4 mask channels into float4 per pixel + per-pixel sq-norms
// ---------------------------------------------------------------------------
__global__ __launch_bounds__(256)
void prep_mask_kernel(const float* __restrict__ mask, float4* __restrict__ mp,
                      float* __restrict__ sq) {
    int idx = blockIdx.x * blockDim.x + threadIdx.x;   // over BATCH*NPIX
    if (idx >= BATCH * NPIX) return;
    int b = idx >> 12;
    int n = idx & (NPIX - 1);
    const float* src = mask + (size_t)b * MCH * NPIX + n;
    float4 v;
    v.x = src[0 * NPIX]; v.y = src[1 * NPIX];
    v.z = src[2 * NPIX]; v.w = src[3 * NPIX];
    mp[idx] = v;
    sq[idx] = v.x * v.x + v.y * v.y + v.z * v.z + v.w * v.w;
}

// ---------------------------------------------------------------------------
// Kernel 3: style-side inverse denominators: 1/(max(||fpatch||,eps)*max(||mpatch||,eps))
// (content-side norms are positive row constants -> argmax invariant, dropped)
// ---------------------------------------------------------------------------
__global__ __launch_bounds__(256)
void invden_kernel(const float* __restrict__ sqf, const float* __restrict__ sqm,
                   float* __restrict__ invden) {
    int idx = blockIdx.x * blockDim.x + threadIdx.x;   // over BATCH*NPIX
    if (idx >= BATCH * NPIX) return;
    int m = idx & (NPIX - 1);
    int mi = m >> 6, mj = m & 63;
    float s1 = 0.f, s2 = 0.f;
#pragma unroll
    for (int di = -1; di <= 1; ++di)
#pragma unroll
        for (int dj = -1; dj <= 1; ++dj) {
            int r = mi + di, c = mj + dj;
            if ((unsigned)r < 64u && (unsigned)c < 64u) {
                int o = di * WW + dj;
                s1 += sqf[idx + o];
                s2 += sqm[idx + o];
            }
        }
    invden[idx] = 1.f / (fmaxf(sqrtf(s1), EPSV) * fmaxf(sqrtf(s2), EPSV));
}

// ---------------------------------------------------------------------------
// Kernel 4: WMMA NT GEMM  G[n][m] = sum_c A[n][c] * B[m][c]   (4096x4096x256)
// 128x128 workgroup tile, 8 wave32 each 64x32 (4x2 of 16x16x32 WMMA).
// ---------------------------------------------------------------------------
#define TILE 128
#define KSTEP 32
#define LDSS 40   // fp16 row stride in LDS (32 data + 8 pad -> conflict-free)

__global__ __launch_bounds__(256)
void gemm_nt_wmma_kernel(const _Float16* __restrict__ A,
                         const _Float16* __restrict__ B,
                         float* __restrict__ G) {
    __shared__ _Float16 Ash[TILE * LDSS];
    __shared__ _Float16 Bsh[TILE * LDSS];

    const int tid   = threadIdx.x;
    const int lane  = tid & 31;
    const int wave  = tid >> 5;
    const int row16 = lane & 15;
    const int sel   = lane >> 4;          // lane-half
    const int waveM = wave & 1;           // 2 waves along M
    const int waveN = wave >> 1;          // 4 waves along N
    const int rowBase = blockIdx.y * TILE;
    const int colBase = blockIdx.x * TILE;

    v8f acc[4][2] = {};

    const int ldr = tid >> 1;             // 0..127 : row within tile
    const int ldh = tid & 1;              // 16-element half of 32-wide K slab

    for (int k0 = 0; k0 < KDIM; k0 += KSTEP) {
        // global -> regs (each thread 32B from A and B)
        const uint4* gA = (const uint4*)(A + (size_t)(rowBase + ldr) * KDIM + k0 + ldh * 16);
        const uint4* gB = (const uint4*)(B + (size_t)(colBase + ldr) * KDIM + k0 + ldh * 16);
        uint4 a0 = gA[0], a1 = gA[1];
        uint4 b0 = gB[0], b1 = gB[1];

        __syncthreads();                  // previous slab fully consumed
        *(uint4*)&Ash[ldr * LDSS + ldh * 16]     = a0;
        *(uint4*)&Ash[ldr * LDSS + ldh * 16 + 8] = a1;
        *(uint4*)&Bsh[ldr * LDSS + ldh * 16]     = b0;
        *(uint4*)&Bsh[ldr * LDSS + ldh * 16 + 8] = b1;
        __syncthreads();

        // A fragments: ISA 16-bit A 16x32 layout -> K chunks {sel*8, 16+sel*8}
        FragU af[4];
#pragma unroll
        for (int mt = 0; mt < 4; ++mt) {
            const _Float16* p = &Ash[(waveM * 64 + mt * 16 + row16) * LDSS + sel * 8];
            af[mt].u[0] = *(const uint4*)p;
            af[mt].u[1] = *(const uint4*)(p + 16);
        }
        // B fragments: same K chunks as A {sel*8, 16+sel*8}
        FragU bf[2];
#pragma unroll
        for (int nt = 0; nt < 2; ++nt) {
            const _Float16* p = &Bsh[(waveN * 32 + nt * 16 + row16) * LDSS + sel * 8];
            bf[nt].u[0] = *(const uint4*)p;
            bf[nt].u[1] = *(const uint4*)(p + 16);
        }
#pragma unroll
        for (int mt = 0; mt < 4; ++mt)
#pragma unroll
            for (int nt = 0; nt < 2; ++nt)
                acc[mt][nt] = __builtin_amdgcn_wmma_f32_16x16x32_f16(
                    false, af[mt].v, false, bf[nt].v,
                    (short)0, acc[mt][nt], false, false);
    }

    // C/D layout: VGPR v -> M = v + 8*sel ; N = lane%16
#pragma unroll
    for (int mt = 0; mt < 4; ++mt)
#pragma unroll
        for (int nt = 0; nt < 2; ++nt)
#pragma unroll
            for (int v = 0; v < 8; ++v) {
                int row = rowBase + waveM * 64 + mt * 16 + v + sel * 8;
                int col = colBase + waveN * 32 + nt * 16 + row16;
                G[(size_t)row * NPIX + col] = acc[mt][nt][v];
            }
}

// ---------------------------------------------------------------------------
// Kernel 5: 9-tap diagonal shift reduce + mask dot + streaming argmax.
// One workgroup per content site n; 256 threads stride over style sites m.
// ---------------------------------------------------------------------------
__global__ __launch_bounds__(256)
void argmax_kernel(const float* __restrict__ G,
                   const float4* __restrict__ cmp,   // content mask pixels (this batch)
                   const float4* __restrict__ smp,   // style mask pixels (this batch)
                   const float* __restrict__ invden, // style denominators (this batch)
                   int* __restrict__ best) {
    __shared__ float sV[256];
    __shared__ int   sI[256];

    const int n  = blockIdx.x;
    const int ni = n >> 6, nj = n & 63;
    const int tid = threadIdx.x;

    int   offs[9], dis[9], djs[9];
    bool  vN[9];
    float4 cm9[9];
#pragma unroll
    for (int d = 0; d < 9; ++d) {
        int di = d / 3 - 1, dj = d % 3 - 1;
        dis[d] = di; djs[d] = dj;
        offs[d] = di * WW + dj;
        int r = ni + di, c = nj + dj;
        vN[d] = ((unsigned)r < 64u) && ((unsigned)c < 64u);
        float4 z; z.x = z.y = z.z = z.w = 0.f;
        cm9[d] = vN[d] ? cmp[n + offs[d]] : z;
    }

    float bV = -INFINITY;
    int   bI = 0x7fffffff;

    for (int m = tid; m < NPIX; m += 256) {
        int mi = m >> 6, mj = m & 63;
        float fd = 0.f, md = 0.f;
#pragma unroll
        for (int d = 0; d < 9; ++d) {
            if (!vN[d]) continue;
            if ((unsigned)(mi + dis[d]) < 64u && (unsigned)(mj + djs[d]) < 64u) {
                int t = m + offs[d];
                fd += G[(size_t)(n + offs[d]) * NPIX + t];   // diagonal-shifted Gram
                float4 sp = smp[t];
                float4 cp = cm9[d];
                md += cp.x * sp.x + cp.y * sp.y + cp.z * sp.z + cp.w * sp.w;
            }
        }
        // hint next chunk of the center Gram row into cache
        if (m + 256 < NPIX)
            __builtin_prefetch(&G[(size_t)n * NPIX + m + 256], 0, 1);
        float sim = fd * md * invden[m];
        if (sim > bV) { bV = sim; bI = m; }
    }

    sV[tid] = bV; sI[tid] = bI;
    __syncthreads();
    for (int s = 128; s > 0; s >>= 1) {
        if (tid < s) {
            if (sV[tid + s] > sV[tid] ||
                (sV[tid + s] == sV[tid] && sI[tid + s] < sI[tid])) {
                sV[tid] = sV[tid + s];
                sI[tid] = sI[tid + s];
            }
        }
        __syncthreads();
    }
    if (tid == 0) best[n] = sI[0];
}

// ---------------------------------------------------------------------------
// Kernel 6: gather winning style patches -> out (B, N, C, 3, 3) fp32
// ---------------------------------------------------------------------------
__global__ __launch_bounds__(256)
void gather_kernel(const float* __restrict__ styleFeat,
                   const int* __restrict__ best,
                   float* __restrict__ out) {
    int idx = blockIdx.x * blockDim.x + threadIdx.x;   // over BATCH*NPIX*CCH
    if (idx >= BATCH * NPIX * CCH) return;
    int b = idx >> 20;                 // NPIX*CCH = 2^20
    int rem = idx & ((1 << 20) - 1);
    int n = rem >> 8;
    int c = rem & (CCH - 1);
    int m = best[b * NPIX + n];
    int mi = m >> 6, mj = m & 63;
    const float* src = styleFeat + ((size_t)b * CCH + c) * NPIX;
    float* o = out + (size_t)idx * 9;
#pragma unroll
    for (int i = 0; i < 3; ++i)
#pragma unroll
        for (int j = 0; j < 3; ++j) {
            int r = mi + i - 1, cc = mj + j - 1;
            o[i * 3 + j] = ((unsigned)r < 64u && (unsigned)cc < 64u)
                               ? src[r * WW + cc] : 0.f;
        }
}

// ---------------------------------------------------------------------------
// Host launcher
// ---------------------------------------------------------------------------
extern "C" void kernel_launch(void* const* d_in, const int* in_sizes, int n_in,
                              void* d_out, int out_size, void* d_ws, size_t ws_size,
                              hipStream_t stream) {
    const float* content_feat = (const float*)d_in[0];
    const float* style_feat   = (const float*)d_in[1];
    const float* content_mask = (const float*)d_in[2];
    const float* style_mask   = (const float*)d_in[3];
    float* out = (float*)d_out;

    // workspace carve-up (256B aligned)
    char* ws = (char*)d_ws;
    size_t off = 0;
    auto carve = [&](size_t bytes) -> char* {
        char* p = ws + off;
        off = (off + bytes + 255) & ~(size_t)255;
        return p;
    };
    const size_t featBytes = (size_t)BATCH * NPIX * KDIM * sizeof(_Float16);
    _Float16* XtC    = (_Float16*)carve(featBytes);
    _Float16* XtS    = (_Float16*)carve(featBytes);
    float*    sqfC   = (float*)carve((size_t)BATCH * NPIX * sizeof(float));
    float*    sqfS   = (float*)carve((size_t)BATCH * NPIX * sizeof(float));
    float*    sqmC   = (float*)carve((size_t)BATCH * NPIX * sizeof(float));
    float*    sqmS   = (float*)carve((size_t)BATCH * NPIX * sizeof(float));
    float4*   cmp    = (float4*)carve((size_t)BATCH * NPIX * sizeof(float4));
    float4*   smp    = (float4*)carve((size_t)BATCH * NPIX * sizeof(float4));
    float*    invden = (float*)carve((size_t)BATCH * NPIX * sizeof(float));
    int*      best   = (int*)carve((size_t)BATCH * NPIX * sizeof(int));
    float*    G      = (float*)carve((size_t)NPIX * NPIX * sizeof(float)); // 64 MB, reused per batch

    const int prepBlocks = (BATCH * NPIX + 255) / 256;     // 32
    prep_feat_kernel<<<prepBlocks, 256, 0, stream>>>(content_feat, XtC, sqfC);
    prep_feat_kernel<<<prepBlocks, 256, 0, stream>>>(style_feat,   XtS, sqfS);
    prep_mask_kernel<<<prepBlocks, 256, 0, stream>>>(content_mask, cmp, sqmC);
    prep_mask_kernel<<<prepBlocks, 256, 0, stream>>>(style_mask,   smp, sqmS);
    invden_kernel<<<prepBlocks, 256, 0, stream>>>(sqfS, sqmS, invden);

    dim3 gemmGrid(NPIX / TILE, NPIX / TILE);               // 32 x 32
    for (int b = 0; b < BATCH; ++b) {
        gemm_nt_wmma_kernel<<<gemmGrid, 256, 0, stream>>>(
            XtC + (size_t)b * NPIX * KDIM,
            XtS + (size_t)b * NPIX * KDIM, G);
        argmax_kernel<<<NPIX, 256, 0, stream>>>(
            G, cmp + (size_t)b * NPIX, smp + (size_t)b * NPIX,
            invden + (size_t)b * NPIX, best + (size_t)b * NPIX);
    }

    const int gatherBlocks = (BATCH * NPIX * CCH + 255) / 256;  // 8192
    gather_kernel<<<gatherBlocks, 256, 0, stream>>>(style_feat, best, out);
}
